// GCNNet_1228360647292
// MI455X (gfx1250) — matrix operand order repaired and checked by reading, not compiled, not run on hardware
//
#include <hip/hip_runtime.h>

// ---------------------------------------------------------------------------
// GCN (2x GCNConv + mean-pool + linear) for gfx1250 (MI455X, wave32, WMMA).
// Dense GEMMs use v_wmma_f32_16x16x32_bf16; sparse aggregation uses
// wave-per-edge float4 gathers + hardware f32 atomics (L2-resident).
// ---------------------------------------------------------------------------

#define DH 128          // D_IN == D_HID == 128
#define NGRAPH 256
#define NCLS 10

typedef __attribute__((ext_vector_type(16))) __bf16 v16bf;
typedef __attribute__((ext_vector_type(8)))  float  v8f;

union Frag16 { v16bf v; uint4 q[2]; };

static __device__ __forceinline__ unsigned short f2bf(float f) {
  unsigned int u = __float_as_uint(f);
  u += 0x7FFFu + ((u >> 16) & 1u);   // round-to-nearest-even
  return (unsigned short)(u >> 16);
}

// ---------------- small elementwise kernels --------------------------------

__global__ void k_fill(float* __restrict__ p, float v, int n) {
  int t = blockIdx.x * blockDim.x + threadIdx.x;
  if (t < n) p[t] = v;
}

__global__ void k_deg_accum(const int* __restrict__ dst, float* __restrict__ deg, int E) {
  int t = blockIdx.x * blockDim.x + threadIdx.x;
  if (t < E) unsafeAtomicAdd(&deg[dst[t]], 1.0f);
}

__global__ void k_rsqrt(float* __restrict__ p, int n) {
  int t = blockIdx.x * blockDim.x + threadIdx.x;
  if (t < n) p[t] = __frsqrt_rn(p[t]);   // deg >= 1 always (self-loop)
}

__global__ void k_norm(const int* __restrict__ src, const int* __restrict__ dst,
                       const float* __restrict__ dinv, float* __restrict__ norm, int E) {
  int t = blockIdx.x * blockDim.x + threadIdx.x;
  if (t < E) norm[t] = dinv[src[t]] * dinv[dst[t]];
}

__global__ void k_f32_to_bf16(const float* __restrict__ x, unsigned short* __restrict__ y, int n) {
  int t = blockIdx.x * blockDim.x + threadIdx.x;
  if (t < n) y[t] = f2bf(x[t]);
}

// Wt[n*128 + k] = bf16(W[k*128 + n])  (128x128)
__global__ void k_transpose_bf16(const float* __restrict__ W, unsigned short* __restrict__ Wt) {
  int t = blockIdx.x * blockDim.x + threadIdx.x;
  if (t < DH * DH) {
    int n = t >> 7, k = t & 127;
    Wt[t] = f2bf(W[k * DH + n]);
  }
}

// agg[i,d] = dinv[i]^2 * h[i,d]   (self-loop contribution, also initializes agg)
__global__ void k_agg_init(const float* __restrict__ dinv, const float* __restrict__ h,
                           float* __restrict__ agg, int n) {
  int t = blockIdx.x * blockDim.x + threadIdx.x;
  if (t < n) {
    float di = dinv[t >> 7];
    agg[t] = di * di * h[t];
  }
}

__global__ void k_bias_relu_bf16(const float* __restrict__ agg, const float* __restrict__ b,
                                 unsigned short* __restrict__ out, int n) {
  int t = blockIdx.x * blockDim.x + threadIdx.x;
  if (t < n) out[t] = f2bf(fmaxf(agg[t] + b[t & 127], 0.0f));
}

__global__ void k_bias_relu_f32(const float* __restrict__ agg, const float* __restrict__ b,
                                float* __restrict__ out, int n) {
  int t = blockIdx.x * blockDim.x + threadIdx.x;
  if (t < n) out[t] = fmaxf(agg[t] + b[t & 127], 0.0f);
}

// ---------------- WMMA GEMM: C[M,128] = A[M,128](bf16) @ W (Bt = W^T, bf16) --

__global__ __launch_bounds__(256) void gcn_gemm_wmma_bf16(
    const unsigned short* __restrict__ A,   // [M,128] bf16 row-major
    const unsigned short* __restrict__ Bt,  // [128,128] bf16, Bt[n][k] = W[k][n]
    float* __restrict__ C, int M)
{
  const int lane = threadIdx.x & 31;
  const int wave = threadIdx.x >> 5;
  const int m0   = blockIdx.x * 128 + wave * 16;   // 16-row strip per wave
  const int half = lane >> 4;
  const int l16  = lane & 15;

  v8f acc[8] = {};   // 8 N-tiles of 16 cols -> full 128-wide output strip

  const int  mrow = m0 + l16;
  const bool mrow_ok = (mrow < M);
  const unsigned short* arow = A + (size_t)mrow * DH;

#pragma unroll
  for (int k = 0; k < 4; ++k) {          // K = 128 in steps of 32
    const int k0 = k * 32;
    Frag16 a;
    if (mrow_ok) {
      // A-frag layout (16-bit, 16x32): elems 0-7 -> K = k0 + half*8 + e,
      //                                elems 8-15 -> K = k0 + 16 + half*8 + e
      const unsigned short* p = arow + k0 + half * 8;
      a.q[0] = *(const uint4*)(p);
      a.q[1] = *(const uint4*)(p + 16);
    } else {
      a.q[0] = make_uint4(0u, 0u, 0u, 0u);
      a.q[1] = make_uint4(0u, 0u, 0u, 0u);
    }
#pragma unroll
    for (int nt = 0; nt < 8; ++nt) {
      // B-frag layout (32x16): lane holds column n, elems 0-7 -> K = k0 + half*8 + e, elems 8-15 -> K = k0 + 16 + half*8 + e
      Frag16 b;
      const unsigned short* pb = Bt + (size_t)(nt * 16 + l16) * DH + k0 + half * 8;
      b.q[0] = *(const uint4*)(pb);
      b.q[1] = *(const uint4*)(pb + 16);
      acc[nt] = __builtin_amdgcn_wmma_f32_16x16x32_bf16(
          false, a.v, false, b.v, (short)0, acc[nt], false, false);
    }
  }

#pragma unroll
  for (int nt = 0; nt < 8; ++nt) {
#pragma unroll
    for (int r = 0; r < 8; ++r) {
      // D layout: lane half selects M base (0 or 8), VGPR r -> +r
      int row = m0 + half * 8 + r;
      if (row < M) C[(size_t)row * DH + nt * 16 + l16] = acc[nt][r];
    }
  }
}

// ---------------- sparse aggregation: wave(32) per edge, float4 per lane ----

__global__ __launch_bounds__(256) void gcn_agg_edges(
    const int* __restrict__ src, const int* __restrict__ dst,
    const float* __restrict__ norm, const float* __restrict__ h,
    float* __restrict__ agg, int E)
{
  int tid  = blockIdx.x * blockDim.x + threadIdx.x;
  int e    = tid >> 5;
  if (e >= E) return;
  int lane = tid & 31;
  int s = src[e], d = dst[e];
  float w = norm[e];
  const float4 v = *(const float4*)(h + (size_t)s * DH + lane * 4);
  float* o = agg + (size_t)d * DH + lane * 4;
  unsafeAtomicAdd(o + 0, v.x * w);
  unsafeAtomicAdd(o + 1, v.y * w);
  unsafeAtomicAdd(o + 2, v.z * w);
  unsafeAtomicAdd(o + 3, v.w * w);
}

// ---------------- pooling + readout ----------------------------------------

__global__ __launch_bounds__(256) void gcn_pool(
    const int* __restrict__ batch, const float* __restrict__ h,
    float* __restrict__ sums, float* __restrict__ cnts, int N)
{
  int tid  = blockIdx.x * blockDim.x + threadIdx.x;
  int i    = tid >> 5;
  if (i >= N) return;
  int lane = tid & 31;
  int g = batch[i];
  const float4 v = *(const float4*)(h + (size_t)i * DH + lane * 4);
  float* o = sums + (size_t)g * DH + lane * 4;
  unsafeAtomicAdd(o + 0, v.x);
  unsafeAtomicAdd(o + 1, v.y);
  unsafeAtomicAdd(o + 2, v.z);
  unsafeAtomicAdd(o + 3, v.w);
  if (lane == 0) unsafeAtomicAdd(&cnts[g], 1.0f);
}

__global__ void gcn_readout(const float* __restrict__ sums, const float* __restrict__ cnts,
                            const float* __restrict__ Wl, const float* __restrict__ bl,
                            float* __restrict__ out)
{
  int t = blockIdx.x * blockDim.x + threadIdx.x;
  if (t >= NGRAPH * NCLS) return;
  int g = t / NCLS, c = t % NCLS;
  float inv = 1.0f / fmaxf(cnts[g], 1.0f);
  float acc = bl[c];
#pragma unroll 4
  for (int d = 0; d < DH; ++d) acc += sums[(size_t)g * DH + d] * inv * Wl[d * NCLS + c];
  out[t] = acc;
}

// ---------------------------------------------------------------------------

static inline size_t alignup(size_t x) { return (x + 255) & ~(size_t)255; }

extern "C" void kernel_launch(void* const* d_in, const int* in_sizes, int n_in,
                              void* d_out, int out_size, void* d_ws, size_t ws_size,
                              hipStream_t stream)
{
  const float* x    = (const float*)d_in[0];
  const int*   edge = (const int*)  d_in[1];   // [2, E]
  const int*   batch= (const int*)  d_in[2];
  const float* W1   = (const float*)d_in[3];
  const float* b1   = (const float*)d_in[4];
  const float* W2   = (const float*)d_in[5];
  const float* b2   = (const float*)d_in[6];
  const float* Wl   = (const float*)d_in[7];
  const float* bl   = (const float*)d_in[8];
  float* out = (float*)d_out;

  const int N = in_sizes[0] / DH;      // 50000
  const int E = in_sizes[1] / 2;       // 800000
  const int* src = edge;
  const int* dst = edge + E;

  // workspace carve-up (~68 MB)
  char* ws = (char*)d_ws;
  size_t off = 0;
  float*          dinv = (float*)(ws + off);          off += alignup((size_t)N * 4);
  float*          norm = (float*)(ws + off);          off += alignup((size_t)E * 4);
  unsigned short* xb   = (unsigned short*)(ws + off); off += alignup((size_t)N * DH * 2);
  float*          h    = (float*)(ws + off);          off += alignup((size_t)N * DH * 4);
  float*          agg  = (float*)(ws + off);          off += alignup((size_t)N * DH * 4);
  unsigned short* w1t  = (unsigned short*)(ws + off); off += alignup((size_t)DH * DH * 2);
  unsigned short* w2t  = (unsigned short*)(ws + off); off += alignup((size_t)DH * DH * 2);
  float*          sums = (float*)(ws + off);          off += alignup((size_t)NGRAPH * DH * 4);
  float*          cnts = (float*)(ws + off);          off += alignup((size_t)NGRAPH * 4);
  (void)ws_size; (void)n_in; (void)out_size;

  const int ND = N * DH;
  auto g1 = [](int n) { return dim3((unsigned)((n + 255) / 256)); };

  // --- normalization: deg (incl. self-loop) -> dinv -> per-edge norm
  k_fill      <<<g1(N), 256, 0, stream>>>(dinv, 1.0f, N);
  k_deg_accum <<<g1(E), 256, 0, stream>>>(dst, dinv, E);
  k_rsqrt     <<<g1(N), 256, 0, stream>>>(dinv, N);
  k_norm      <<<g1(E), 256, 0, stream>>>(src, dst, dinv, norm, E);

  // --- bf16 staging of x and transposed weights
  k_f32_to_bf16   <<<g1(ND), 256, 0, stream>>>(x,  xb,  ND);
  k_transpose_bf16<<<g1(DH * DH), 256, 0, stream>>>(W1, w1t);
  k_transpose_bf16<<<g1(DH * DH), 256, 0, stream>>>(W2, w2t);

  const int gemm_blocks = (N + 127) / 128;

  // --- layer 1: h = x @ W1 ; agg = A_norm @ h ; relu(+b1) -> bf16 (reuse xb)
  gcn_gemm_wmma_bf16<<<gemm_blocks, 256, 0, stream>>>(xb, w1t, h, N);
  k_agg_init  <<<g1(ND), 256, 0, stream>>>(dinv, h, agg, ND);
  gcn_agg_edges<<<g1(E * 32), 256, 0, stream>>>(src, dst, norm, h, agg, E);
  k_bias_relu_bf16<<<g1(ND), 256, 0, stream>>>(agg, b1, xb, ND);

  // --- layer 2: h = h1 @ W2 ; agg = A_norm @ h ; relu(+b2) -> f32 (into h)
  gcn_gemm_wmma_bf16<<<gemm_blocks, 256, 0, stream>>>(xb, w2t, h, N);
  k_agg_init  <<<g1(ND), 256, 0, stream>>>(dinv, h, agg, ND);
  gcn_agg_edges<<<g1(E * 32), 256, 0, stream>>>(src, dst, norm, h, agg, E);
  k_bias_relu_f32<<<g1(ND), 256, 0, stream>>>(agg, b2, h, ND);

  // --- global mean pool + linear head
  k_fill  <<<g1(NGRAPH * DH), 256, 0, stream>>>(sums, 0.0f, NGRAPH * DH);
  k_fill  <<<g1(NGRAPH), 256, 0, stream>>>(cnts, 0.0f, NGRAPH);
  gcn_pool<<<g1(N * 32), 256, 0, stream>>>(batch, h, sums, cnts, N);
  gcn_readout<<<g1(NGRAPH * NCLS), 256, 0, stream>>>(sums, cnts, Wl, bl, out);
}
